// MTCGCNN_49323404427326
// MI455X (gfx1250) — matrix operand order repaired and checked by reading, not compiled, not run on hardware
//
#include <hip/hip_runtime.h>

// ---------------- problem constants (from reference) ----------------
#define NATOM 131072
#define MNBR  12
#define ORIGF 92
#define NBRF  41
#define AF    64
#define C2    128   // 2*AF
#define FIN   169   // 2*AF + NBRF
#define KP    192   // FIN padded to multiple of 32
#define KE    96    // ORIGF padded to multiple of 32
#define NCONV 3
#define N0    1024
#define NP    2
#define HH    128
#define EPSB  1e-5f

typedef __attribute__((ext_vector_type(16))) __bf16 v16bf;
typedef __attribute__((ext_vector_type(8)))  float  v8f;
typedef __attribute__((ext_vector_type(4)))  unsigned int v4u;

union Frag { v16bf v; unsigned short s[16]; v4u q[2]; };

__device__ __forceinline__ unsigned short f2bf(float f) {
  union { float f; unsigned int u; } cv; cv.f = f;
  unsigned int u = cv.u;
  u += 0x7FFFu + ((u >> 16) & 1u);     // round-to-nearest-even
  return (unsigned short)(u >> 16);
}
__device__ __forceinline__ float softplusf(float x) {
  return x > 20.f ? x : log1pf(expf(x));
}
__device__ __forceinline__ float sigmoidf(float x) {
  return 1.f / (1.f + expf(-x));
}

// ---------------- weight prep: fp32 -> bf16, fragment-major swizzle --------
// Conv weights:  Wpad[l][kc][t][lane][j]  (kc<6, t<8, lane<32, j<16)
//   value = W[k][c] with k = kc*32 + (lane>>4)*16 + j, c = t*16 + (lane&15)
// Embed weights: Wepad[kc][t][lane][j]    (kc<3, t<4)
__global__ void prep_weights(const float* __restrict__ convW,
                             const float* __restrict__ Wemb,
                             unsigned short* __restrict__ Wpad,
                             unsigned short* __restrict__ Wepad) {
  int i = blockIdx.x * blockDim.x + threadIdx.x;
  const int tot1 = NCONV * KP * C2;  // 73728
  if (i < tot1) {
    int l = i / (KP * C2); int rem = i % (KP * C2);
    int kc = rem >> 12;           // / 4096
    int rem2 = rem & 4095;
    int t = rem2 >> 9;            // / 512
    int rem3 = rem2 & 511;
    int lane = rem3 >> 4, j = rem3 & 15;
    int k = kc * 32 + ((lane >> 4) << 4) + j;
    int c = t * 16 + (lane & 15);
    Wpad[i] = (k < FIN) ? f2bf(convW[(l * FIN + k) * C2 + c]) : (unsigned short)0;
  } else {
    int jj = i - tot1;
    if (jj < KE * AF) {           // 6144
      int kc = jj >> 11;          // / 2048
      int rem2 = jj & 2047;
      int t = rem2 >> 9;
      int rem3 = rem2 & 511;
      int lane = rem3 >> 4, j = rem3 & 15;
      int k = kc * 32 + ((lane >> 4) << 4) + j;
      int c = t * 16 + (lane & 15);
      Wepad[jj] = (k < ORIGF) ? f2bf(Wemb[k * AF + c]) : (unsigned short)0;
    }
  }
}

// ---------------- embedding GEMM: x = atom_fea @ W_embed + b (bf16 WMMA) ----
__global__ __launch_bounds__(256) void embed_gemm(
    const float* __restrict__ atom_fea, const unsigned short* __restrict__ Wepad,
    const float* __restrict__ b_embed, float* __restrict__ x,
    unsigned short* __restrict__ xbf) {
  __shared__ alignas(16) unsigned short lW[KE * AF];      // 12 KB (swizzled)
  __shared__ alignas(16) unsigned short lA[8][16 * KE];   // 24 KB
  int tid = threadIdx.x;
  for (int i = tid; i < (KE * AF) / 8; i += 256)
    ((v4u*)lW)[i] = ((const v4u*)Wepad)[i];
  int wave = tid >> 5, lane = tid & 31;
  int mrow = lane & 15, kh = lane >> 4, n = lane & 15;
  int rowBase = blockIdx.x * 128 + wave * 16;
  { // build A tile (16 rows x 96 K, bf16), pair-packed stores
    for (int i = lane; i < 16 * 48; i += 32) {
      int row = i / 48, cp = i % 48;
      const float* src = atom_fea + (size_t)(rowBase + row) * ORIGF;
      int c0 = 2 * cp, c1 = c0 + 1;
      unsigned int v0 = (c0 < ORIGF) ? f2bf(src[c0]) : 0u;
      unsigned int v1 = (c1 < ORIGF) ? f2bf(src[c1]) : 0u;
      ((unsigned int*)&lA[wave][row * KE])[cp] = v0 | (v1 << 16);
    }
  }
  __syncthreads();
  v8f acc[4] = {};
  const v4u* Aq = (const v4u*)(&lA[wave][0]);
  const v4u* Wq = (const v4u*)lW;
#pragma unroll
  for (int kc = 0; kc < 3; ++kc) {
    Frag fa;
    int ab = mrow * 12 + kc * 4 + 2 * kh;   // KE row pitch = 12 v4u
    fa.q[0] = Aq[ab]; fa.q[1] = Aq[ab + 1];
#pragma unroll
    for (int t = 0; t < 4; ++t) {
      Frag fb;
      int bb = ((kc * 4 + t) * 32 + lane) * 2;
      fb.q[0] = Wq[bb]; fb.q[1] = Wq[bb + 1];
      acc[t] = __builtin_amdgcn_wmma_f32_16x16x32_bf16(
          false, fa.v, false, fb.v, (short)0, acc[t], false, false);
    }
  }
#pragma unroll
  for (int t = 0; t < 4; ++t) {
    int col = t * 16 + n;
    float bias = b_embed[col];
#pragma unroll
    for (int e = 0; e < 8; ++e) {
      int ga = rowBase + e + 8 * kh;
      float v = acc[t][e] + bias;
      x[(size_t)ga * AF + col] = v;
      xbf[(size_t)ga * AF + col] = f2bf(v);
    }
  }
}

// ---------------- conv GEMM (two-pass: MODE 0 = stats, MODE 1 = apply) -----
template <int MODE>
__global__ __launch_bounds__(256) void conv_pass(
    const unsigned short* __restrict__ xbf, const float* __restrict__ nbr_fea,
    const int* __restrict__ nbr_idx, const unsigned short* __restrict__ WpadL,
    const float* __restrict__ convbL, float* __restrict__ stats1,
    const float* __restrict__ g1, const float* __restrict__ be1,
    float* __restrict__ nbr_sumed) {
  __shared__ alignas(16) unsigned short lW[KP * C2];      // 48 KB (swizzled)
  __shared__ alignas(16) unsigned short lA[8][16 * KP];   // 48 KB
  __shared__ float sred[2 * C2];                          // 1 KB (stats pass)
  int tid = threadIdx.x;
  for (int i = tid; i < (KP * C2) / 8; i += 256)
    ((v4u*)lW)[i] = ((const v4u*)WpadL)[i];
  int wave = tid >> 5, lane = tid & 31;
  int mrow = lane & 15, kh = lane >> 4, n = lane & 15;
  long rowBase = (long)blockIdx.x * 128 + wave * 16;
  { // build A tile: [self x(64) | nbr x(64) | edge(41) | pad(23)] as bf16
    v4u* dstq = (v4u*)(&lA[wave][0]);              // row pitch = 24 v4u
    const v4u* xq = (const v4u*)xbf;
    for (int i = lane; i < 128; i += 32) {         // self rows, 16B vectors
      int row = i >> 3, c4 = i & 7;
      int aat = (int)((rowBase + row) / MNBR);
      dstq[row * 24 + c4] = xq[(size_t)aat * 8 + c4];
    }
    for (int i = lane; i < 128; i += 32) {         // gathered neighbor rows
      int row = i >> 3, c4 = i & 7;
      int nb = nbr_idx[rowBase + row];
      dstq[row * 24 + 8 + c4] = xq[(size_t)nb * 8 + c4];
    }
    for (int i = lane; i < 16 * 32; i += 32) {     // edge feats, pair-packed
      int row = i >> 5, cp = i & 31;
      const float* nf = nbr_fea + (size_t)(rowBase + row) * NBRF;
      int c0 = 2 * cp, c1 = c0 + 1;
      unsigned int v0 = (c0 < NBRF) ? f2bf(nf[c0]) : 0u;
      unsigned int v1 = (c1 < NBRF) ? f2bf(nf[c1]) : 0u;
      ((unsigned int*)&lA[wave][row * KP + 2 * AF])[cp] = v0 | (v1 << 16);
    }
  }
  if (MODE == 0) { for (int i = tid; i < 2 * C2; i += 256) sred[i] = 0.f; }
  __syncthreads();

  v8f acc[8] = {};
  const v4u* Aq = (const v4u*)(&lA[wave][0]);
  const v4u* Wq = (const v4u*)lW;
#pragma unroll
  for (int kc = 0; kc < 6; ++kc) {
    Frag fa;
    int ab = mrow * 24 + kc * 4 + 2 * kh;   // KP row pitch = 24 v4u
    fa.q[0] = Aq[ab]; fa.q[1] = Aq[ab + 1];
#pragma unroll
    for (int t = 0; t < 8; ++t) {
      Frag fb;
      int bb = ((kc * 8 + t) * 32 + lane) * 2;
      fb.q[0] = Wq[bb]; fb.q[1] = Wq[bb + 1];
      acc[t] = __builtin_amdgcn_wmma_f32_16x16x32_bf16(
          false, fa.v, false, fb.v, (short)0, acc[t], false, false);
    }
  }

  if (MODE == 0) { // per-column sum / sumsq of gated (bias included)
#pragma unroll
    for (int t = 0; t < 8; ++t) {
      int col = t * 16 + n;
      float bias = convbL[col];
      float s = 0.f, q = 0.f;
#pragma unroll
      for (int e = 0; e < 8; ++e) { float v = acc[t][e] + bias; s += v; q += v * v; }
      atomicAdd(&sred[col], s);
      atomicAdd(&sred[C2 + col], q);
    }
    __syncthreads();
    for (int i = tid; i < 2 * C2; i += 256) atomicAdd(&stats1[i], sred[i]);
  } else {       // BN1 + sigmoid*softplus + reduce over M into nbr_sumed
    const float* meanA = stats1 + 2 * C2;
    const float* invA  = stats1 + 3 * C2;
    float mf[4], sf[4], tf[4], bf_[4], mc[4], sc[4], tc[4], bc[4];
#pragma unroll
    for (int t = 0; t < 4; ++t) {
      int cf = t * 16 + n, cc = cf + AF;
      bf_[t] = convbL[cf];  bc[t] = convbL[cc];
      mf[t] = meanA[cf];    mc[t] = meanA[cc];
      sf[t] = invA[cf] * g1[cf]; sc[t] = invA[cc] * g1[cc];
      tf[t] = be1[cf];      tc[t] = be1[cc];
    }
    // rows handled by this lane are 8 consecutive (atoms change every 12):
    // accumulate per-atom runs in registers, flush with <=2 atomic bursts
    long r0 = rowBase + 8 * kh;
    int aat = (int)(r0 / MNBR);
    float accv[4] = {0.f, 0.f, 0.f, 0.f};
#pragma unroll
    for (int e = 0; e < 8; ++e) {
      int an = (int)((r0 + e) / MNBR);
      if (an != aat) {
        float* dst = nbr_sumed + (size_t)aat * AF;
#pragma unroll
        for (int t = 0; t < 4; ++t) {
          atomicAdd(&dst[t * 16 + n], accv[t]);
          accv[t] = 0.f;
        }
        aat = an;
      }
#pragma unroll
      for (int t = 0; t < 4; ++t) {
        float gf = (acc[t][e]     + bf_[t] - mf[t]) * sf[t] + tf[t];
        float gc = (acc[t + 4][e] + bc[t]  - mc[t]) * sc[t] + tc[t];
        accv[t] += sigmoidf(gf) * softplusf(gc);
      }
    }
    float* dst = nbr_sumed + (size_t)aat * AF;
#pragma unroll
    for (int t = 0; t < 4; ++t) atomicAdd(&dst[t * 16 + n], accv[t]);
  }
}

// ---------------- BN stat finalize: mean & inv-std from sum/sumsq ----------
__global__ void finalize_stats(float* st, int width, float invcnt) {
  int c = blockIdx.x * blockDim.x + threadIdx.x;
  if (c < width) {
    float mean = st[c] * invcnt;
    float var = st[width + c] * invcnt - mean * mean;
    st[2 * width + c] = mean;
    st[3 * width + c] = rsqrtf(var + EPSB);
  }
}

// ---------------- column sums over nbr_sumed (N x 64) for BN2 --------------
__global__ __launch_bounds__(256) void col_stats64(const float* __restrict__ data,
                                                   float* __restrict__ st2) {
  __shared__ float sS[AF], sQ[AF];
  int tid = threadIdx.x;
  if (tid < AF) { sS[tid] = 0.f; sQ[tid] = 0.f; }
  __syncthreads();
  int col = tid & 63, sub = tid >> 6;
  const long rows_per_blk = NATOM / 256;
  long r0 = (long)blockIdx.x * rows_per_blk;
  float s = 0.f, q = 0.f;
  for (long r = r0 + sub; r < r0 + rows_per_blk; r += 4) {
    float v = data[r * AF + col];
    s += v; q += v * v;
  }
  atomicAdd(&sS[col], s); atomicAdd(&sQ[col], q);
  __syncthreads();
  if (tid < AF) { atomicAdd(&st2[tid], sS[tid]); atomicAdd(&st2[AF + tid], sQ[tid]); }
}

// ---------------- x = softplus(x + BN2(nbr_sumed)) -------------------------
__global__ void update_x(float* __restrict__ x, unsigned short* __restrict__ xbf,
                         const float* __restrict__ nbr_sumed,
                         const float* __restrict__ st2,
                         const float* __restrict__ g2, const float* __restrict__ be2) {
  long i = (long)blockIdx.x * blockDim.x + threadIdx.x;
  if (i >= (long)NATOM * AF) return;
  int f = (int)(i & 63);
  float bn = g2[f] * (nbr_sumed[i] - st2[2 * AF + f]) * st2[3 * AF + f] + be2[f];
  float v = softplusf(x[i] + bn);
  x[i] = v; xbf[i] = f2bf(v);
}

// ---------------- segment-mean pooling -------------------------------------
__global__ void pool_accum(const float* __restrict__ x, const int* __restrict__ cidx,
                           float* __restrict__ sums, float* __restrict__ counts) {
  long i = (long)blockIdx.x * blockDim.x + threadIdx.x;
  if (i >= (long)NATOM * AF) return;
  int a = (int)(i >> 6), f = (int)(i & 63);
  int c = cidx[a];
  atomicAdd(&sums[(size_t)c * AF + f], x[i]);
  if (f == 0) atomicAdd(&counts[c], 1.f);
}
__global__ void crys_finalize(const float* __restrict__ sums,
                              const float* __restrict__ counts,
                              float* __restrict__ crys_out, float* __restrict__ crys_a) {
  int i = blockIdx.x * blockDim.x + threadIdx.x;
  if (i >= N0 * AF) return;
  int c = i >> 6;
  float cnt = counts[c]; cnt = cnt > 1.f ? cnt : 1.f;
  float v = sums[i] / cnt;
  crys_out[i] = v;
  crys_a[i] = softplusf(v);
}

// ---------------- 2-head MLP on 1024 crystals (tiny, fp32) -----------------
__global__ void head_kernel(const float* __restrict__ a, const float* __restrict__ W1,
                            const float* __restrict__ b1, const float* __restrict__ W2,
                            const float* __restrict__ b2, float* __restrict__ out) {
  int t = blockIdx.x * blockDim.x + threadIdx.x;
  if (t >= N0 * NP) return;
  int p = t & 1, c = t >> 1;
  const float* av = a + (size_t)c * AF;
  const float* w1 = W1 + (size_t)p * AF * HH;
  const float* bb1 = b1 + p * HH;
  const float* w2 = W2 + p * HH;
  float o = 0.f;
  for (int h = 0; h < HH; ++h) {
    float acc = bb1[h];
    for (int f = 0; f < AF; ++f) acc += av[f] * w1[f * HH + h];
    o += softplusf(acc) * w2[h];
  }
  out[c * NP + p] = o + b2[p];
}

// ---------------- orchestration --------------------------------------------
extern "C" void kernel_launch(void* const* d_in, const int* in_sizes, int n_in,
                              void* d_out, int out_size, void* d_ws, size_t ws_size,
                              hipStream_t stream) {
  const float* atom_fea = (const float*)d_in[0];
  const float* nbr_fea  = (const float*)d_in[1];
  const float* W_embed  = (const float*)d_in[2];
  const float* b_embed  = (const float*)d_in[3];
  const float* conv_W   = (const float*)d_in[4];
  const float* conv_b   = (const float*)d_in[5];
  const float* conv_g1  = (const float*)d_in[6];
  const float* conv_be1 = (const float*)d_in[7];
  const float* conv_g2  = (const float*)d_in[8];
  const float* conv_be2 = (const float*)d_in[9];
  const float* head_W1  = (const float*)d_in[10];
  const float* head_b1  = (const float*)d_in[11];
  const float* head_W2  = (const float*)d_in[12];
  const float* head_b2  = (const float*)d_in[13];
  const int* nbr_idx    = (const int*)d_in[14];
  const int* crys_idx   = (const int*)d_in[15];

  char* ws = (char*)d_ws;
  float* x            = (float*)(ws + 0);                 // 33,554,432 B
  unsigned short* xbf = (unsigned short*)(ws + 33554432); // 16,777,216 B
  float* nbr_sumed    = (float*)(ws + 50331648);          // 33,554,432 B
  unsigned short* Wpad  = (unsigned short*)(ws + 83886080); // 147,456 B
  unsigned short* Wepad = (unsigned short*)(ws + 84033536); //  12,288 B
  float* stats1 = (float*)(ws + 84045824);                // 512 floats
  float* stats2 = (float*)(ws + 84047872);                // 256 floats
  float* psum   = (float*)(ws + 84048896);                // N0*AF floats
  float* pcnt   = (float*)(ws + 84311040);                // N0 floats
  float* crysa  = (float*)(ws + 84315136);                // N0*AF floats
  float* outp = (float*)d_out;
  float* crys_out = outp + N0 * NP;

  const int totW = NCONV * KP * C2 + KE * AF;
  prep_weights<<<(totW + 255) / 256, 256, 0, stream>>>(conv_W, W_embed, Wpad, Wepad);
  embed_gemm<<<NATOM / 128, 256, 0, stream>>>(atom_fea, Wepad, b_embed, x, xbf);

  const long NM = (long)NATOM * MNBR;
  for (int l = 0; l < NCONV; ++l) {
    hipMemsetAsync(stats1, 0, 2 * C2 * sizeof(float), stream);
    conv_pass<0><<<(int)(NM / 128), 256, 0, stream>>>(
        xbf, nbr_fea, nbr_idx, Wpad + (size_t)l * KP * C2, conv_b + l * C2,
        stats1, nullptr, nullptr, nullptr);
    finalize_stats<<<1, 128, 0, stream>>>(stats1, C2, 1.f / (float)NM);
    hipMemsetAsync(nbr_sumed, 0, (size_t)NATOM * AF * sizeof(float), stream);
    conv_pass<1><<<(int)(NM / 128), 256, 0, stream>>>(
        xbf, nbr_fea, nbr_idx, Wpad + (size_t)l * KP * C2, conv_b + l * C2,
        stats1, conv_g1 + l * C2, conv_be1 + l * C2, nbr_sumed);
    hipMemsetAsync(stats2, 0, 2 * AF * sizeof(float), stream);
    col_stats64<<<256, 256, 0, stream>>>(nbr_sumed, stats2);
    finalize_stats<<<1, 64, 0, stream>>>(stats2, AF, 1.f / (float)NATOM);
    update_x<<<(NATOM * AF) / 256, 256, 0, stream>>>(
        x, xbf, nbr_sumed, stats2, conv_g2 + l * AF, conv_be2 + l * AF);
  }

  hipMemsetAsync(psum, 0, (size_t)N0 * AF * sizeof(float), stream);
  hipMemsetAsync(pcnt, 0, N0 * sizeof(float), stream);
  pool_accum<<<(NATOM * AF) / 256, 256, 0, stream>>>(x, crys_idx, psum, pcnt);
  crys_finalize<<<(N0 * AF) / 256, 256, 0, stream>>>(psum, pcnt, crys_out, crysa);
  head_kernel<<<(N0 * NP + 255) / 256, 256, 0, stream>>>(
      crysa, head_W1, head_b1, head_W2, head_b2, outp);
}
